// Golu_56702158242365
// MI455X (gfx1250) — compile-verified, matrix operand order checked by reading, not run on hardware
//
#include <hip/hip_runtime.h>
#include <hip/hip_bf16.h>

#define DEVINL __device__ __forceinline__

typedef __attribute__((ext_vector_type(16))) __bf16 v16bf;
typedef __attribute__((ext_vector_type(8)))  float  v8f;
typedef int v4i_ __attribute__((vector_size(16)));

// ---- model constants (from reference) ----
constexpr int   B_  = 2, T_ = 512, E_ = 768, NH_ = 12, HD_ = 64, V_ = 256, P_ = 4, L_ = 4;
constexpr float SCALE_ = 0.125f;              // HD^-0.5
constexpr int   M1_ = B_ * T_;                // 1024 token rows
constexpr int   M2_ = B_ * T_ * P_;           // 4096 sa-block rows
constexpr int   E3_ = 3 * E_;                 // 2304

// ---------------------------------------------------------------------------
// f32 -> bf16 (round to nearest even)
// ---------------------------------------------------------------------------
DEVINL unsigned short f2bf(float f) {
  unsigned int u = __float_as_uint(f);
  u += 0x7FFFu + ((u >> 16) & 1u);
  return (unsigned short)(u >> 16);
}

// ---------------------------------------------------------------------------
// CDNA5 async global->LDS copy (16B per lane), with sync fallback.
// Builtin signature (from clang diagnostic): (v4i AS1*, v4i AS3*, i32, i32)
// ---------------------------------------------------------------------------
#if __has_builtin(__builtin_amdgcn_global_load_async_to_lds_b128)
#define HAVE_ASYNC_LDS 1
#endif

DEVINL void copy16_g2l(unsigned short* lds, const unsigned short* g) {
#ifdef HAVE_ASYNC_LDS
  __builtin_amdgcn_global_load_async_to_lds_b128(
      (__attribute__((address_space(1))) v4i_*)(g),
      (__attribute__((address_space(3))) v4i_*)(lds), 0, 0);
#else
  *(uint4*)lds = *(const uint4*)g;
#endif
}
DEVINL void wait_async_copies() {
#ifdef HAVE_ASYNC_LDS
#if __has_builtin(__builtin_amdgcn_s_wait_asynccnt)
  __builtin_amdgcn_s_wait_asynccnt(0);
#else
  asm volatile("s_wait_asynccnt 0" ::: "memory");
#endif
#endif
}

union FragU { uint4 u[2]; v16bf v; };

// A/B 16-bit fragment for v_wmma_f32_16x16x32_bf16: lane (l%16 = row/col,
// l/16 = half) holds K = 8*half + [0..7] and K = 16 + 8*half + [0..7]
// -> two contiguous 16-byte LDS reads when tiles are stored K-contiguous.
DEVINL v16bf load_frag_lds(const unsigned short* rowbase, int hf) {
  FragU f;
  f.u[0] = *(const uint4*)(rowbase + hf * 8);
  f.u[1] = *(const uint4*)(rowbase + 16 + hf * 8);
  return f.v;
}

// ---------------------------------------------------------------------------
// Weight convert+transpose: in f32 [batch][K][N] -> out bf16 [batch][N][K].
// One-time pass so GEMMs stream bf16 K-contiguous tiles for both operands.
// ---------------------------------------------------------------------------
__global__ void convt_kernel(const float* __restrict__ in, unsigned short* __restrict__ out,
                             int K, int N, long long total) {
  long long idx = (long long)blockIdx.x * 256 + threadIdx.x;
  if (idx >= total) return;
  long long kn = (long long)K * N;
  long long l = idx / kn, rem = idx - l * kn;
  int k = (int)(rem / N), n = (int)(rem - (long long)k * N);
  out[l * kn + (long long)n * K + k] = f2bf(in[idx]);
}

// ---------------------------------------------------------------------------
// Tiled GEMM: C[M,N] = epi(A[M,K] @ W[N,K]^T), A and W bf16, f32 accumulate.
// Block tile 64x128, 8 waves, each wave 32x32 via 4 WMMAs / 32-wide K-step.
// EPI bits: 1 = +bias[n], 2 = exact GELU, 4 = +R[m,n] f32 residual,
//           8 = store bf16 (else f32).
// ---------------------------------------------------------------------------
constexpr int BM = 64, BN = 128, BK = 32;
constexpr int LDT = BK + 8;   // ushort stride; row starts stay 16B aligned

template <int EPI>
__global__ __launch_bounds__(256)
void gemm_bf16_kernel(const unsigned short* __restrict__ A, const unsigned short* __restrict__ W,
                      const float* __restrict__ bias, const float* __restrict__ R,
                      void* __restrict__ Cout, int M, int N, int K) {
  __shared__ __align__(16) unsigned short As[BM * LDT];
  __shared__ __align__(16) unsigned short Bs[BN * LDT];

  const int tid  = threadIdx.x;
  const int lane = tid & 31;
  const int wid  = tid >> 5;
  const int bm   = blockIdx.y * BM;
  const int bn   = blockIdx.x * BN;
  const int wm   = (wid >> 2) * 32;    // 0,32
  const int wn   = (wid & 3) * 32;     // 0,32,64,96
  const int hf   = lane >> 4;
  const int lrow = lane & 15;

  v8f c00 = {}, c01 = {}, c10 = {}, c11 = {};

  // cooperative staging: each thread moves one (A) + two (B) 16-byte chunks
  const int sr = tid >> 2;             // 0..63
  const int sc = (tid & 3) * 8;        // 0,8,16,24

  for (int k0 = 0; k0 < K; k0 += BK) {
    copy16_g2l(&As[sr * LDT + sc], A + (size_t)(bm + sr) * K + k0 + sc);
    copy16_g2l(&Bs[sr * LDT + sc],        W + (size_t)(bn + sr) * K + k0 + sc);
    copy16_g2l(&Bs[(sr + 64) * LDT + sc], W + (size_t)(bn + sr + 64) * K + k0 + sc);
    wait_async_copies();
    __syncthreads();

    if (k0 + BK < K) {  // prefetch next tiles -> global_prefetch_b8
      __builtin_prefetch(A + (size_t)(bm + sr) * K + (k0 + BK) + sc, 0, 0);
      __builtin_prefetch(W + (size_t)(bn + sr) * K + (k0 + BK) + sc, 0, 0);
      __builtin_prefetch(W + (size_t)(bn + sr + 64) * K + (k0 + BK) + sc, 0, 0);
    }

    v16bf a0 = load_frag_lds(&As[(wm + lrow) * LDT], hf);
    v16bf a1 = load_frag_lds(&As[(wm + 16 + lrow) * LDT], hf);
    v16bf b0 = load_frag_lds(&Bs[(wn + lrow) * LDT], hf);
    v16bf b1 = load_frag_lds(&Bs[(wn + 16 + lrow) * LDT], hf);

    c00 = __builtin_amdgcn_wmma_f32_16x16x32_bf16(false, a0, false, b0, (short)0, c00, false, false);
    c01 = __builtin_amdgcn_wmma_f32_16x16x32_bf16(false, a0, false, b1, (short)0, c01, false, false);
    c10 = __builtin_amdgcn_wmma_f32_16x16x32_bf16(false, a1, false, b0, (short)0, c10, false, false);
    c11 = __builtin_amdgcn_wmma_f32_16x16x32_bf16(false, a1, false, b1, (short)0, c11, false, false);
    __syncthreads();
  }

  // ---- epilogue: C/D layout: VGPR r -> M = 8*half + r, lane%16 -> N ----
  float* Cf = (float*)Cout;
  unsigned short* Ch = (unsigned short*)Cout;
  auto store_tile = [&](v8f acc, int tm, int tn) {
#pragma unroll
    for (int r = 0; r < 8; ++r) {
      const int m = bm + tm + hf * 8 + r;
      const int n = bn + tn + lrow;
      float v = acc[r];
      if (EPI & 1) v += bias[n];
      if (EPI & 2) v = 0.5f * v * (1.0f + erff(v * 0.70710678118654752f));
      if (EPI & 4) v += R[(size_t)m * N + n];
      if (EPI & 8) Ch[(size_t)m * N + n] = f2bf(v);
      else         Cf[(size_t)m * N + n] = v;
    }
  };
  store_tile(c00, wm, wn);
  store_tile(c01, wm, wn + 16);
  store_tile(c10, wm + 16, wn);
  store_tile(c11, wm + 16, wn + 16);
}

// ---------------------------------------------------------------------------
// Embedding gather: x0 = x = emb[tokens]
// ---------------------------------------------------------------------------
__global__ void embed_kernel(const int* __restrict__ tokens, const float* __restrict__ emb,
                             float* __restrict__ x0, float* __restrict__ x) {
  int idx = blockIdx.x * blockDim.x + threadIdx.x;
  if (idx >= M1_ * E_) return;
  int row = idx / E_, e = idx - row * E_;
  float v = emb[(size_t)tokens[row] * E_ + e];
  x0[idx] = v;
  x[idx]  = v;
}

// ---------------------------------------------------------------------------
// LayerNorm over D (biased variance, eps 1e-5); bf16 output feeds GEMM A.
// ---------------------------------------------------------------------------
__global__ __launch_bounds__(256)
void ln_kernel(const float* __restrict__ x, const float* __restrict__ g,
               const float* __restrict__ b, unsigned short* __restrict__ y, int D) {
  const int row = blockIdx.x, tid = threadIdx.x;
  const float* xr = x + (size_t)row * D;
  unsigned short* yr = y + (size_t)row * D;
  __shared__ float s1[256], s2[256];
  float a = 0.f, q = 0.f;
  for (int i = tid; i < D; i += 256) { float v = xr[i]; a += v; q += v * v; }
  s1[tid] = a; s2[tid] = q; __syncthreads();
  for (int off = 128; off > 0; off >>= 1) {
    if (tid < off) { s1[tid] += s1[tid + off]; s2[tid] += s2[tid + off]; }
    __syncthreads();
  }
  const float mean = s1[0] / D;
  const float var  = s2[0] / D - mean * mean;
  const float inv  = rsqrtf(var + 1e-5f);
  for (int i = tid; i < D; i += 256) yr[i] = f2bf((xr[i] - mean) * inv * g[i] + b[i]);
}

// ---------------------------------------------------------------------------
// In-place RoPE on q and k slices of a [rows, 3E] f32 qkv buffer.
// position = row % mod (mod = T for token stream, P for sa block).
// ---------------------------------------------------------------------------
__global__ void rope_kernel(float* __restrict__ buf, int rows, int mod) {
  int idx = blockIdx.x * blockDim.x + threadIdx.x;
  int total = rows * NH_ * 32;
  if (idx >= total) return;
  int d = idx & 31;
  int h = (idx >> 5) % NH_;
  int r = idx / (NH_ * 32);
  float pos = (float)(r % mod);
  float ang = pos * __powf(10000.0f, -(float)d / 32.0f);
  float sn, cs; __sincosf(ang, &sn, &cs);
  size_t base = (size_t)r * E3_ + h * HD_ + d;
  float x1 = buf[base], x2 = buf[base + 32];
  buf[base]      = x1 * cs - x2 * sn;
  buf[base + 32] = x1 * sn + x2 * cs;
  size_t kb = base + E_;
  float y1 = buf[kb], y2 = buf[kb + 32];
  buf[kb]      = y1 * cs - y2 * sn;
  buf[kb + 32] = y1 * sn + y2 * cs;
}

// ---------------------------------------------------------------------------
// Sliding-window causal attention: query qt attends keys max(0,qt-S+1)..qt.
// Adds result into x.  grid (NH, T, B), 64 threads; S <= 128.
// ---------------------------------------------------------------------------
__global__ __launch_bounds__(64)
void patch_attn_kernel(const float* __restrict__ qkv, float* __restrict__ x, int S) {
  const int h = blockIdx.x, qt = blockIdx.y, b = blockIdx.z, tid = threadIdx.x;
  __shared__ float qs[HD_];
  __shared__ float scr[128];
  const size_t qrow = ((size_t)(b * T_ + qt)) * E3_ + h * HD_;
  qs[tid] = qkv[qrow + tid];
  __syncthreads();
  int ks = qt - S + 1; if (ks < 0) ks = 0;
  const int nk = qt - ks + 1;
  for (int i = tid; i < nk; i += 64) {
    const float* kr = qkv + ((size_t)(b * T_ + ks + i)) * E3_ + E_ + h * HD_;
    float s = 0.f;
#pragma unroll 8
    for (int d = 0; d < HD_; ++d) s += qs[d] * kr[d];
    scr[i] = s * SCALE_;
  }
  __syncthreads();
  float m = -3.0e38f;
  for (int j = 0; j < nk; ++j) m = fmaxf(m, scr[j]);
  __syncthreads();
  for (int i = tid; i < nk; i += 64) scr[i] = __expf(scr[i] - m);
  __syncthreads();
  float ssum = 0.f;
  for (int j = 0; j < nk; ++j) ssum += scr[j];
  float o = 0.f;
  for (int j = 0; j < nk; ++j)
    o += scr[j] * qkv[((size_t)(b * T_ + ks + j)) * E3_ + 2 * E_ + h * HD_ + tid];
  x[((size_t)(b * T_ + qt)) * E_ + h * HD_ + tid] += o / ssum;
}

// ---------------------------------------------------------------------------
// SA-block attention over P=4 patches per (b,t,head); no mask.
// Writes bf16 o[(b,t,p), h*64+d] (feeds the proj GEMM as A).
// ---------------------------------------------------------------------------
__global__ __launch_bounds__(64)
void sa_attn_kernel(const float* __restrict__ qkv, unsigned short* __restrict__ o) {
  const int h = blockIdx.x, t = blockIdx.y, b = blockIdx.z, tid = threadIdx.x;
  __shared__ float qs[P_][HD_], ks_[P_][HD_], vs[P_][HD_], sc[P_][P_];
  const size_t base = ((size_t)(b * T_ + t)) * P_;
#pragma unroll
  for (int p = 0; p < P_; ++p) {
    size_t r3 = (base + p) * E3_ + h * HD_;
    qs[p][tid]  = qkv[r3 + tid];
    ks_[p][tid] = qkv[r3 + E_ + tid];
    vs[p][tid]  = qkv[r3 + 2 * E_ + tid];
  }
  __syncthreads();
  if (tid < 16) {
    int i = tid >> 2, j = tid & 3;
    float s = 0.f;
#pragma unroll 8
    for (int d = 0; d < HD_; ++d) s += qs[i][d] * ks_[j][d];
    sc[i][j] = s * SCALE_;
  }
  __syncthreads();
  if (tid < 4) {
    float m = fmaxf(fmaxf(sc[tid][0], sc[tid][1]), fmaxf(sc[tid][2], sc[tid][3]));
    float e0 = __expf(sc[tid][0] - m), e1 = __expf(sc[tid][1] - m);
    float e2 = __expf(sc[tid][2] - m), e3 = __expf(sc[tid][3] - m);
    float su = e0 + e1 + e2 + e3;
    sc[tid][0] = e0 / su; sc[tid][1] = e1 / su; sc[tid][2] = e2 / su; sc[tid][3] = e3 / su;
  }
  __syncthreads();
#pragma unroll
  for (int i = 0; i < P_; ++i) {
    float acc = sc[i][0] * vs[0][tid] + sc[i][1] * vs[1][tid] +
                sc[i][2] * vs[2][tid] + sc[i][3] * vs[3][tid];
    o[(base + i) * E_ + h * HD_ + tid] = f2bf(acc);
  }
}

// ---------------------------------------------------------------------------
__global__ void add_kernel(float* __restrict__ x, const float* __restrict__ a, int n) {
  int i = blockIdx.x * blockDim.x + threadIdx.x;
  if (i < n) x[i] += a[i];
}

__global__ void slice_kernel(const float* __restrict__ sa_x, float* __restrict__ out) {
  int i = blockIdx.x * blockDim.x + threadIdx.x;
  if (i >= M1_ * V_) return;
  int row = i / V_, v = i - row * V_;
  out[i] = sa_x[(size_t)row * P_ * V_ + v];   // p == 0 slice
}

// ---------------------------------------------------------------------------
extern "C" void kernel_launch(void* const* d_in, const int* in_sizes, int n_in,
                              void* d_out, int out_size, void* d_ws, size_t ws_size,
                              hipStream_t stream) {
  (void)in_sizes; (void)n_in; (void)out_size; (void)ws_size;
  const int*   tokens    = (const int*)d_in[0];
  const float* emb       = (const float*)d_in[1];
  const float* bl0g      = (const float*)d_in[2];
  const float* bl0b      = (const float*)d_in[3];
  const float* bl1g      = (const float*)d_in[4];
  const float* bl1b      = (const float*)d_in[5];
  const float* bqkv      = (const float*)d_in[6];
  const float* bw1       = (const float*)d_in[7];
  const float* bb1       = (const float*)d_in[8];
  const float* bw2       = (const float*)d_in[9];
  const float* bb2       = (const float*)d_in[10];
  const float* fing      = (const float*)d_in[11];
  const float* finb      = (const float*)d_in[12];
  const float* head_w    = (const float*)d_in[13];
  const float* head_b    = (const float*)d_in[14];
  const float* sa_l0g    = (const float*)d_in[15];
  const float* sa_l0b    = (const float*)d_in[16];
  const float* sa_l1g    = (const float*)d_in[17];
  const float* sa_l1b    = (const float*)d_in[18];
  const float* sa_qkv_w  = (const float*)d_in[19];
  const float* sa_proj_w = (const float*)d_in[20];
  const float* sa_proj_b = (const float*)d_in[21];
  const float* sa_w1     = (const float*)d_in[22];
  const float* sa_b1     = (const float*)d_in[23];
  const float* sa_w2     = (const float*)d_in[24];
  const float* sa_b2     = (const float*)d_in[25];
  float* out = (float*)d_out;

  // ---- workspace carve-out ----
  char* wsp = (char*)d_ws;
  auto alloc = [&](size_t bytes) { char* p = wsp; wsp += (bytes + 255) & ~(size_t)255; return p; };
  float* x0      = (float*)alloc((size_t)M1_ * E_ * 4);
  float* x       = (float*)alloc((size_t)M1_ * E_ * 4);
  float* qkv     = (float*)alloc((size_t)M1_ * E3_ * 4);
  float* logits  = (float*)alloc((size_t)M2_ * V_ * 4);
  float* sa_qkv  = (float*)alloc((size_t)M2_ * E3_ * 4);
  float* sa_x    = (float*)alloc((size_t)M2_ * V_ * 4);
  unsigned short* h_bf     = (unsigned short*)alloc((size_t)M1_ * E_ * 2);
  unsigned short* ff_bf    = (unsigned short*)alloc((size_t)M1_ * 4 * E_ * 2);
  unsigned short* sa_h_bf  = (unsigned short*)alloc((size_t)M2_ * V_ * 2);
  unsigned short* sa_ff_bf = (unsigned short*)alloc((size_t)M2_ * 4 * V_ * 2);
  unsigned short* sa_o_bf  = (unsigned short*)alloc((size_t)M2_ * E_ * 2);
  // bf16 transposed weights [N][K]
  unsigned short* wqkv   = (unsigned short*)alloc((size_t)L_ * E_ * E3_ * 2);
  unsigned short* wff1   = (unsigned short*)alloc((size_t)L_ * E_ * 4 * E_ * 2);
  unsigned short* wff2   = (unsigned short*)alloc((size_t)L_ * 4 * E_ * E_ * 2);
  unsigned short* whead  = (unsigned short*)alloc((size_t)E_ * V_ * P_ * 2);
  unsigned short* wsaqkv = (unsigned short*)alloc((size_t)V_ * E3_ * 2);
  unsigned short* wsapj  = (unsigned short*)alloc((size_t)E_ * V_ * 2);
  unsigned short* wsaf1  = (unsigned short*)alloc((size_t)V_ * 4 * V_ * 2);
  unsigned short* wsaf2  = (unsigned short*)alloc((size_t)4 * V_ * V_ * 2);

  auto convt = [&](const float* in, unsigned short* o_, int K, int N, int batch) {
    long long tot = (long long)batch * K * N;
    convt_kernel<<<(unsigned)((tot + 255) / 256), 256, 0, stream>>>(in, o_, K, N, tot);
  };
  auto ggrid = [](int M, int N) { return dim3((unsigned)(N / BN), (unsigned)(M / BM)); };

  // ---- one-time weight convert + transpose to bf16 [N][K] ----
  convt(bqkv, wqkv, E_, E3_, L_);
  convt(bw1, wff1, E_, 4 * E_, L_);
  convt(bw2, wff2, 4 * E_, E_, L_);
  convt(head_w, whead, E_, V_ * P_, 1);
  convt(sa_qkv_w, wsaqkv, V_, E3_, 1);
  convt(sa_proj_w, wsapj, E_, V_, 1);
  convt(sa_w1, wsaf1, V_, 4 * V_, 1);
  convt(sa_w2, wsaf2, 4 * V_, V_, 1);

  // ---- embedding ----
  embed_kernel<<<(M1_ * E_ + 255) / 256, 256, 0, stream>>>(tokens, emb, x0, x);

  // ---- 4 transformer layers ----
  for (int i = 0; i < L_; ++i) {
    const int S = 16 << i;  // window 16,32,64,128
    ln_kernel<<<M1_, 256, 0, stream>>>(x, bl0g + (size_t)i * E_, bl0b + (size_t)i * E_, h_bf, E_);
    gemm_bf16_kernel<0><<<ggrid(M1_, E3_), 256, 0, stream>>>(
        h_bf, wqkv + (size_t)i * E_ * E3_, nullptr, nullptr, qkv, M1_, E3_, E_);
    rope_kernel<<<(M1_ * NH_ * 32 + 255) / 256, 256, 0, stream>>>(qkv, M1_, T_);
    patch_attn_kernel<<<dim3(NH_, T_, B_), 64, 0, stream>>>(qkv, x, S);
    ln_kernel<<<M1_, 256, 0, stream>>>(x, bl1g + (size_t)i * E_, bl1b + (size_t)i * E_, h_bf, E_);
    gemm_bf16_kernel<11><<<ggrid(M1_, 4 * E_), 256, 0, stream>>>(        // bias+GELU, bf16 out
        h_bf, wff1 + (size_t)i * E_ * 4 * E_, bb1 + (size_t)i * 4 * E_, nullptr, ff_bf, M1_, 4 * E_, E_);
    gemm_bf16_kernel<5><<<ggrid(M1_, E_), 256, 0, stream>>>(             // bias+residual, f32 out
        ff_bf, wff2 + (size_t)i * 4 * E_ * E_, bb2 + (size_t)i * E_, x, x, M1_, E_, 4 * E_);
  }

  // ---- final residual + LN + head ----
  add_kernel<<<(M1_ * E_ + 255) / 256, 256, 0, stream>>>(x, x0, M1_ * E_);
  ln_kernel<<<M1_, 256, 0, stream>>>(x, fing, finb, h_bf, E_);
  gemm_bf16_kernel<1><<<ggrid(M1_, V_ * P_), 256, 0, stream>>>(
      h_bf, whead, head_b, nullptr, logits, M1_, V_ * P_, E_);

  // ---- sa block over (B*T*P, V) rows ----
  ln_kernel<<<M2_, 256, 0, stream>>>(logits, sa_l0g, sa_l0b, sa_h_bf, V_);
  gemm_bf16_kernel<0><<<ggrid(M2_, E3_), 256, 0, stream>>>(
      sa_h_bf, wsaqkv, nullptr, nullptr, sa_qkv, M2_, E3_, V_);
  rope_kernel<<<(M2_ * NH_ * 32 + 255) / 256, 256, 0, stream>>>(sa_qkv, M2_, P_);
  sa_attn_kernel<<<dim3(NH_, T_, B_), 64, 0, stream>>>(sa_qkv, sa_o_bf);
  gemm_bf16_kernel<5><<<ggrid(M2_, V_), 256, 0, stream>>>(               // bias+residual(logits)
      sa_o_bf, wsapj, sa_proj_b, logits, sa_x, M2_, V_, E_);
  ln_kernel<<<M2_, 256, 0, stream>>>(sa_x, sa_l1g, sa_l1b, sa_h_bf, V_);
  gemm_bf16_kernel<11><<<ggrid(M2_, 4 * V_), 256, 0, stream>>>(          // bias+GELU, bf16 out
      sa_h_bf, wsaf1, sa_b1, nullptr, sa_ff_bf, M2_, 4 * V_, V_);
  gemm_bf16_kernel<5><<<ggrid(M2_, V_), 256, 0, stream>>>(               // bias+residual
      sa_ff_bf, wsaf2, sa_b2, sa_x, sa_x, M2_, V_, 4 * V_);

  // ---- output: p == 0 slice ----
  slice_kernel<<<(M1_ * V_ + 255) / 256, 256, 0, stream>>>(sa_x, out);
}
